// VectorQuantizer_16896401342955
// MI455X (gfx1250) — compile-verified, matrix operand order checked by reading, not run on hardware
//
#include <hip/hip_runtime.h>
#include <hip/hip_bf16.h>

// ---------------------------------------------------------------------------
// VQ-VAE vector quantizer for MI455X (gfx1250), wave32 + WMMA bf16.
//   N = 16384 vectors, D = 512, K = 8192 codes.
//   Pass 1: codebook f32 -> bf16 + ||c||^2
//   Pass 2: fused GEMM(x, cb^T) + running argmin, B tiles staged in LDS via
//           global_load_async_to_lds_b128 (double-buffered, ASYNCcnt),
//           math on V_WMMA_F32_16X16X32_BF16, A (16 rows x 512) in registers.
//   Pass 3: gather q = cb[code], loss sum, segment sums via atomics
//   Pass 4/5: EMA updates + Laplace smoothing + final codebook
// ---------------------------------------------------------------------------

#define K_CODES 8192
#define DIMV    512
#define N_VEC   16384
#define ROWS_PER_BLOCK 128      // 8 waves * 16 rows
#define TILE_J  32              // codes per LDS tile
#define LDS_ROW_STRIDE 520      // bf16 elems: 512 + 8 pad -> 260 dwords, 260%64=4
#define TILE_ELEMS (TILE_J * LDS_ROW_STRIDE)
#define EMA_DECAY_F 0.99f
#define EPS_F       1e-5f

typedef __attribute__((ext_vector_type(16))) __bf16 v16bf;
typedef __attribute__((ext_vector_type(8)))  float  v8f;

union FragU { v16bf v; uint4 q[2]; };

__device__ __forceinline__ float bf2f(unsigned short u) {
    unsigned int t = ((unsigned int)u) << 16;
    float f; __builtin_memcpy(&f, &t, 4); return f;
}
__device__ __forceinline__ unsigned short f2bf(float f) {
    unsigned int t; __builtin_memcpy(&t, &f, 4);
    unsigned int r = t + 0x7FFFu + ((t >> 16) & 1u);   // round-to-nearest-even
    return (unsigned short)(r >> 16);
}

// Global-side fragment load (used for the register-resident A matrix).
// ISA 16-bit operand layout (7.12.2): lanes 0-15 hold K={0..7,16..23} of the
// 32-chunk, lanes 16-31 hold K={8..15,24..31}; two contiguous 16B loads.
__device__ __forceinline__ v16bf load_frag16(const unsigned short* __restrict__ rowPtr,
                                             int dChunk, int hi) {
    FragU f;
    const int base = dChunk * 32 + (hi ? 8 : 0);
    f.q[0] = *reinterpret_cast<const uint4*>(rowPtr + base);
    f.q[1] = *reinterpret_cast<const uint4*>(rowPtr + base + 16);
    return f.v;
}

// LDS-side fragment load for B (codebook row m of the staged tile).
__device__ __forceinline__ v16bf lds_frag16(const unsigned short* sbuf,
                                            int m, int dChunk, int hi) {
    FragU f;
    const int base = dChunk * 32 + (hi ? 8 : 0);
    const unsigned short* p = sbuf + m * LDS_ROW_STRIDE + base;
    f.q[0] = *reinterpret_cast<const uint4*>(p);       // ds_load_b128
    f.q[1] = *reinterpret_cast<const uint4*>(p + 16);  // ds_load_b128
    return f.v;
}

// Cooperative async prefetch of one 32x512 bf16 codebook tile into LDS.
// 2048 chunks of 16B, 256 threads -> 8 async issues per thread (ASYNCcnt +8).
__device__ __forceinline__ void prefetch_tile(const unsigned short* __restrict__ cb_bf,
                                              int j0, unsigned short* sbuf, int tid) {
#pragma unroll
    for (int i = 0; i < 8; ++i) {
        const int c    = tid + i * 256;
        const int row  = c >> 6;          // 64 x 16B chunks per 512-elem row
        const int col  = (c & 63) << 3;   // bf16 element offset
        const unsigned short* g = cb_bf + (size_t)(j0 + row) * DIMV + col;
        const unsigned lds = (unsigned)(uintptr_t)(sbuf + row * LDS_ROW_STRIDE + col);
        asm volatile("global_load_async_to_lds_b128 %0, %1, off"
                     :: "v"(lds), "v"(g) : "memory");
    }
}

// ---------------------------------------------------------------------------
// Pass 1: codebook f32 -> bf16, and ||c||^2. One block per code row.
// ---------------------------------------------------------------------------
__global__ __launch_bounds__(256)
void vq_prep_codebook(const float* __restrict__ cb,
                      unsigned short* __restrict__ cb_bf,
                      float* __restrict__ cnorm) {
    const int k = blockIdx.x;
    const float* row = cb + (size_t)k * DIMV;
    unsigned short* orow = cb_bf + (size_t)k * DIMV;
    float s = 0.f;
    for (int d = threadIdx.x; d < DIMV; d += 256) {
        float v = row[d];
        s += v * v;
        orow[d] = f2bf(v);
    }
    __shared__ float red[8];
    for (int m = 16; m >= 1; m >>= 1) s += __shfl_xor(s, m, 32);
    if ((threadIdx.x & 31) == 0) red[threadIdx.x >> 5] = s;
    __syncthreads();
    if (threadIdx.x < 8) {
        s = red[threadIdx.x];
        for (int m = 4; m >= 1; m >>= 1) s += __shfl_xor(s, m, 32);
        if (threadIdx.x == 0) cnorm[k] = s;
    }
}

// ---------------------------------------------------------------------------
// Pass 2: fused bf16 WMMA GEMM + argmin with LDS double-buffered B tiles.
// Block = 8 waves * 16 rows = 128 rows; argmin_k(||c||^2 - 2 x.c).
// ---------------------------------------------------------------------------
__global__ __launch_bounds__(256)
void vq_argmin_wmma(const unsigned short* __restrict__ x_bf,
                    const unsigned short* __restrict__ cb_bf,
                    const float* __restrict__ cnorm,
                    int* __restrict__ codes) {
    __shared__ unsigned short smem[2 * TILE_ELEMS];   // 2 x 33.3 KB

    const int tid  = threadIdx.x;
    const int lane = tid & 31;
    const int wave = tid >> 5;
    const int m    = lane & 15;       // A row / B code column within tile
    const int hi   = lane >> 4;
    const int rowBase = blockIdx.x * ROWS_PER_BLOCK + wave * 16;

    // Register-resident A: this wave's 16 rows x full D=512 (16 chunks of 32).
    const unsigned short* xrow = x_bf + (size_t)(rowBase + m) * DIMV;
    v16bf a[16];
#pragma unroll
    for (int kc = 0; kc < 16; ++kc) a[kc] = load_frag16(xrow, kc, hi);

    float best[8];
    int   bidx[8];
#pragma unroll
    for (int r = 0; r < 8; ++r) { best[r] = 3.0e38f; bidx[r] = 0; }

    prefetch_tile(cb_bf, 0, smem, tid);
    int buf = 0;

    for (int j0 = 0; j0 < K_CODES; j0 += TILE_J) {
        if (j0 + TILE_J < K_CODES) {
            prefetch_tile(cb_bf, j0 + TILE_J, smem + (buf ^ 1) * TILE_ELEMS, tid);
            asm volatile("s_wait_asynccnt 0x8" ::: "memory");  // tile j0 landed
        } else {
            asm volatile("s_wait_asynccnt 0x0" ::: "memory");
        }
        __syncthreads();   // all waves' slices of tile j0 are in LDS

        const unsigned short* sb = smem + buf * TILE_ELEMS;
        v8f acc0 = {};
        v8f acc1 = {};
#pragma unroll
        for (int kc = 0; kc < 16; ++kc) {
            v16bf b0 = lds_frag16(sb, m,      kc, hi);
            v16bf b1 = lds_frag16(sb, m + 16, kc, hi);
            acc0 = __builtin_amdgcn_wmma_f32_16x16x32_bf16(
                false, a[kc], false, b0, (short)0, acc0, false, false);
            acc1 = __builtin_amdgcn_wmma_f32_16x16x32_bf16(
                false, a[kc], false, b1, (short)0, acc1, false, false);
        }

        const float cn0 = cnorm[j0 + m];
        const float cn1 = cnorm[j0 + 16 + m];
        const int   j0i = j0 + m;
        const int   j1i = j0 + 16 + m;
#pragma unroll
        for (int r = 0; r < 8; ++r) {
            float d0 = __builtin_fmaf(-2.0f, acc0[r], cn0);
            float d1 = __builtin_fmaf(-2.0f, acc1[r], cn1);
            if (d0 < best[r]) { best[r] = d0; bidx[r] = j0i; }
            if (d1 < best[r]) { best[r] = d1; bidx[r] = j1i; }
        }

        __syncthreads();   // everyone done reading buf before it is refilled
        buf ^= 1;
    }

    // Min-reduce over the 16 code columns (lanes within each 16-lane half).
#pragma unroll
    for (int mask = 1; mask < 16; mask <<= 1) {
#pragma unroll
        for (int r = 0; r < 8; ++r) {
            float ov = __shfl_xor(best[r], mask, 32);
            int   oi = __shfl_xor(bidx[r], mask, 32);
            if (ov < best[r] || (ov == best[r] && oi < bidx[r])) {
                best[r] = ov; bidx[r] = oi;
            }
        }
    }
    if (m == 0) {  // lane 0 -> rows 0..7, lane 16 -> rows 8..15
#pragma unroll
        for (int r = 0; r < 8; ++r) codes[rowBase + hi * 8 + r] = bidx[r];
    }
}

// ---------------------------------------------------------------------------
// Pass 3: gather quantized, loss sum, segment sums (counts, embed_sum).
// One block per vector row.
// ---------------------------------------------------------------------------
__global__ __launch_bounds__(256)
void vq_scatter_stats(const unsigned short* __restrict__ x_bf,
                      const float* __restrict__ cb,
                      const int* __restrict__ codes,
                      float* __restrict__ out_q,
                      float* __restrict__ out_codes,
                      float* __restrict__ counts,
                      float* __restrict__ embed_sum,
                      float* __restrict__ loss_acc) {
    const int n = blockIdx.x;
    const int k = codes[n];
    const unsigned short* xr = x_bf + (size_t)n * DIMV;
    const float* cr = cb + (size_t)k * DIMV;
    float* qr = out_q + (size_t)n * DIMV;
    float* er = embed_sum + (size_t)k * DIMV;
    float ls = 0.f;
    for (int d = threadIdx.x; d < DIMV; d += 256) {
        float xv = bf2f(xr[d]);
        float qv = cr[d];
        float df = qv - xv;
        ls += df * df;
        qr[d] = qv;                  // straight-through value == q
        atomicAdd(&er[d], xv);       // segment_sum(x)
    }
    __shared__ float red[8];
    for (int m = 16; m >= 1; m >>= 1) ls += __shfl_xor(ls, m, 32);
    if ((threadIdx.x & 31) == 0) red[threadIdx.x >> 5] = ls;
    __syncthreads();
    if (threadIdx.x == 0) {
        float s = 0.f;
        for (int w = 0; w < 8; ++w) s += red[w];
        atomicAdd(loss_acc, s);
        atomicAdd(&counts[k], 1.0f);
        out_codes[n] = (float)k;
    }
}

// ---------------------------------------------------------------------------
// Pass 4: updated_cluster_size + total n.
// ---------------------------------------------------------------------------
__global__ __launch_bounds__(256)
void vq_ema_cluster(const float* __restrict__ ema_cs,
                    const float* __restrict__ counts,
                    float* __restrict__ out_ucs,
                    float* __restrict__ n_acc) {
    const int k = blockIdx.x * 256 + threadIdx.x;
    float ucs = EMA_DECAY_F * ema_cs[k] + (1.0f - EMA_DECAY_F) * counts[k];
    out_ucs[k] = ucs;
    __shared__ float red[8];
    float s = ucs;
    for (int m = 16; m >= 1; m >>= 1) s += __shfl_xor(s, m, 32);
    if ((threadIdx.x & 31) == 0) red[threadIdx.x >> 5] = s;
    __syncthreads();
    if (threadIdx.x == 0) {
        float t = 0.f;
        for (int w = 0; w < 8; ++w) t += red[w];
        atomicAdd(n_acc, t);
    }
}

// ---------------------------------------------------------------------------
// Pass 5: updated_embed_sum, smoothed codebook, losses.
// ---------------------------------------------------------------------------
__global__ __launch_bounds__(256)
void vq_finalize(const float* __restrict__ ema_es,
                 const float* __restrict__ embed_sum,
                 const float* __restrict__ ucs,
                 const float* __restrict__ scal,      // [0]=loss_sum, [1]=n
                 float* __restrict__ out_ucb,
                 float* __restrict__ out_ues,
                 float* __restrict__ out_losses) {
    const size_t kd = (size_t)blockIdx.x * 256 + threadIdx.x;
    const int k = (int)(kd / DIMV);
    const float n = scal[1];
    float ues = EMA_DECAY_F * ema_es[kd] + (1.0f - EMA_DECAY_F) * embed_sum[kd];
    float smoothed = (ucs[k] + EPS_F) / (n + (float)K_CODES * EPS_F) * n;
    out_ues[kd] = ues;
    out_ucb[kd] = ues / smoothed;
    if (kd == 0) {
        float mloss = scal[0] / (float)((size_t)N_VEC * DIMV);
        out_losses[0] = mloss;   // commitment_loss (values identical)
        out_losses[1] = mloss;   // codebook_loss
    }
}

// ---------------------------------------------------------------------------
extern "C" void kernel_launch(void* const* d_in, const int* in_sizes, int n_in,
                              void* d_out, int out_size, void* d_ws, size_t ws_size,
                              hipStream_t stream) {
    const unsigned short* x_bf  = (const unsigned short*)d_in[0]; // bf16 bits
    const float*          cb    = (const float*)d_in[1];
    const float*          emacs = (const float*)d_in[2];
    const float*          emaes = (const float*)d_in[3];
    float* out = (float*)d_out;

    // Output layout (floats, reference return order):
    const size_t OFF_Q     = 0;
    const size_t OFF_CODES = OFF_Q + (size_t)N_VEC * DIMV;       // 8388608
    const size_t OFF_LOSS  = OFF_CODES + N_VEC;                  // +16384
    const size_t OFF_UCB   = OFF_LOSS + 2;
    const size_t OFF_UCS   = OFF_UCB + (size_t)K_CODES * DIMV;
    const size_t OFF_UES   = OFF_UCS + K_CODES;

    // Workspace layout (bytes):
    char* ws = (char*)d_ws;
    unsigned short* cb_bf = (unsigned short*)ws;                          // 8 MiB
    float* cnorm  = (float*)(ws + (size_t)8 * 1024 * 1024);               // 32 KiB
    int*   codes  = (int*)  (ws + (size_t)8 * 1024 * 1024 + 32 * 1024);   // 64 KiB
    float* counts = (float*)(ws + (size_t)8 * 1024 * 1024 + 96 * 1024);   // 32 KiB
    float* esum   = (float*)(ws + (size_t)8 * 1024 * 1024 + 128 * 1024);  // 16 MiB
    float* scal   = (float*)(ws + (size_t)24 * 1024 * 1024 + 128 * 1024); // 8 B

    // Zero the accumulation buffers every call (deterministic, capture-legal).
    hipMemsetAsync(counts, 0, K_CODES * sizeof(float), stream);
    hipMemsetAsync(esum,   0, (size_t)K_CODES * DIMV * sizeof(float), stream);
    hipMemsetAsync(scal,   0, 2 * sizeof(float), stream);

    vq_prep_codebook<<<K_CODES, 256, 0, stream>>>(cb, cb_bf, cnorm);
    vq_argmin_wmma<<<N_VEC / ROWS_PER_BLOCK, 256, 0, stream>>>(x_bf, cb_bf, cnorm, codes);
    vq_scatter_stats<<<N_VEC, 256, 0, stream>>>(x_bf, cb, codes,
                                                out + OFF_Q, out + OFF_CODES,
                                                counts, esum, scal);
    vq_ema_cluster<<<K_CODES / 256, 256, 0, stream>>>(emacs, counts, out + OFF_UCS, scal + 1);
    vq_finalize<<<(size_t)K_CODES * DIMV / 256, 256, 0, stream>>>(
        emaes, esum, out + OFF_UCS, scal,
        out + OFF_UCB, out + OFF_UES, out + OFF_LOSS);
}
